// DoubleSubstitutionHeadAutoRegressive_61220463837577
// MI455X (gfx1250) — compile-verified, matrix operand order checked by reading, not run on hardware
//
#include <hip/hip_runtime.h>
#include <stdint.h>

// ---------------------------------------------------------------------------
// DoubleSubstitutionHeadAutoRegressive for MI455X (gfx1250, wave32, WMMA)
// B=8 L=8192 E=256 CS=8 V=16 M=1024
// All GEMMs on v_wmma_f32_16x16x32_bf16 (f32 accumulate). ~52 GFLOP total;
// ~180MB of intermediates -> L2-resident (192MB / 23.3TB/s HBM).
// Strict-causal block prefix fused into the WMMA C operand (t[7] never done).
// GEMM tiling: 64 A-rows per WG staged in LDS (32KB of 320KB), 8 waves each
// owning 32 N-columns -> 8 accumulators/wave, 4x B-fragment reuse, weight
// (L2) traffic ~0.75GB total so the WMMA pipe, not L2, is the limiter.
// ---------------------------------------------------------------------------

#define BB   8
#define LTOK 8192
#define EE   256
#define CSZ  8
#define MM   1024
#define VV   16
#define MT   64         // A-rows per workgroup in GEMM kernels

typedef __bf16 bf16_t;
typedef bf16_t v16bf __attribute__((ext_vector_type(16)));
typedef float  v8f   __attribute__((ext_vector_type(8)));

union Frag  { v16bf v; uint4 u[2]; };
union Pack8 { uint4 u; unsigned short s[8]; };

__device__ __forceinline__ unsigned short f2bf(float f) {
  union { float f; unsigned int u; } x; x.f = f;
  return (unsigned short)((x.u + 0x7FFFu + ((x.u >> 16) & 1u)) >> 16);
}
__device__ __forceinline__ float bf2f(unsigned short h) {
  union { float f; unsigned int u; } x; x.u = ((unsigned int)h) << 16;
  return x.f;
}

// A/B fragment: lane m=lane&15; lanes<16 hold K=c0..c0+7 and c0+16..c0+23 with
// c0=0, lanes>=16 with c0=8 (CDNA5 16-bit 16x32 operand layout).
__device__ __forceinline__ Frag load_frag(const unsigned short* p) {
  Frag f;
  f.u[0] = *(const uint4*)(p);
  f.u[1] = *(const uint4*)(p + 16);
  return f;
}

__device__ __forceinline__ v8f wmma_bf16(const Frag& a, const Frag& b, v8f c) {
  return __builtin_amdgcn_wmma_f32_16x16x32_bf16(false, a.v, false, b.v,
                                                 (short)0, c, false, false);
}

// ---------------------------------------------------------------------------
// Mask scan: rank (cumsum-1) per position + compacted true-position list.
// ---------------------------------------------------------------------------
__global__ __launch_bounds__(256) void build_index_kernel(
    const int* __restrict__ value,
    int* __restrict__ rank1, int* __restrict__ idx1,
    int* __restrict__ rank2, int* __restrict__ idx2)
{
  const int b   = blockIdx.x;
  const int lev = blockIdx.y;                 // 0 -> mask1 (val_1), 1 -> mask2 (val_2)
  const int* vb = value + (size_t)b * 3 * LTOK + (lev == 0 ? LTOK : 0);
  int* rank = (lev == 0 ? rank1 : rank2) + (size_t)b * LTOK;
  int* idx  = (lev == 0 ? idx1  : idx2 ) + (size_t)b * MM;

  __shared__ int cnts[256];
  const int t = threadIdx.x;
  const int base = t * 32;
  int c = 0;
  for (int i = 0; i < 32; ++i) c += (vb[base + i] == 2) ? 1 : 0;
  cnts[t] = c;
  __syncthreads();
  for (int ofs = 1; ofs < 256; ofs <<= 1) {       // inclusive scan
    int vprev = (t >= ofs) ? cnts[t - ofs] : 0;
    __syncthreads();
    cnts[t] += vprev;
    __syncthreads();
  }
  int run = (t > 0) ? cnts[t - 1] : 0;            // exclusive prefix
  for (int i = 0; i < 32; ++i) {
    int m = (vb[base + i] == 2) ? 1 : 0;
    if (m) idx[run] = base + i;
    run += m;
    int rk = run - 1;
    if (rk < 0) rk = 0; if (rk > MM - 1) rk = MM - 1;
    rank[base + i] = rk;
  }
}

// ---------------------------------------------------------------------------
// Weight transpose + f32->bf16:  dst[j][n][k] = src[j*sj + k*sk + n*sn]
// ---------------------------------------------------------------------------
__global__ __launch_bounds__(256) void wtcvt_kernel(
    const float* __restrict__ src, unsigned short* __restrict__ dst,
    int J, int K, int N, long sj, long sk, long sn)
{
  long e = (long)blockIdx.x * 256 + threadIdx.x;
  long total = (long)J * K * N;
  if (e >= total) return;
  long j = e / ((long)K * N);
  long r = e % ((long)K * N);
  long n = r / K;
  long k = r % K;
  dst[e] = f2bf(src[j * sj + k * sk + n * sn]);
}

// ---------------------------------------------------------------------------
// Embedding lookup -> bf16 rows. one wave per token row (512B).
// ---------------------------------------------------------------------------
__global__ __launch_bounds__(256) void embed_kernel(
    const float* __restrict__ table, const int* __restrict__ value,
    unsigned short* __restrict__ h)
{
  const int wave = threadIdx.x >> 5, lane = threadIdx.x & 31;
  const long row = (long)blockIdx.x * 8 + wave;          // 0..B*L-1
  const long b = row / LTOK, t = row % LTOK;
  const int tok = value[b * 3 * LTOK + 2 * LTOK + t];
  const float4* sp = (const float4*)(table + (size_t)tok * EE) + lane * 2;
  float4 f0 = sp[0], f1 = sp[1];
  Pack8 p;
  p.s[0] = f2bf(f0.x); p.s[1] = f2bf(f0.y); p.s[2] = f2bf(f0.z); p.s[3] = f2bf(f0.w);
  p.s[4] = f2bf(f1.x); p.s[5] = f2bf(f1.y); p.s[6] = f2bf(f1.z); p.s[7] = f2bf(f1.w);
  ((uint4*)(h + row * EE))[lane] = p.u;
}

// f32 rows -> bf16 rows (for x). one wave per row.
__global__ __launch_bounds__(256) void cvt_rows_kernel(
    const float* __restrict__ src, unsigned short* __restrict__ dst)
{
  const int wave = threadIdx.x >> 5, lane = threadIdx.x & 31;
  const long row = (long)blockIdx.x * 8 + wave;
  const float4* sp = (const float4*)(src + row * EE) + lane * 2;
  float4 f0 = sp[0], f1 = sp[1];
  Pack8 p;
  p.s[0] = f2bf(f0.x); p.s[1] = f2bf(f0.y); p.s[2] = f2bf(f0.z); p.s[3] = f2bf(f0.w);
  p.s[4] = f2bf(f1.x); p.s[5] = f2bf(f1.y); p.s[6] = f2bf(f1.z); p.s[7] = f2bf(f1.w);
  ((uint4*)(dst + row * EE))[lane] = p.u;
}

// ---------------------------------------------------------------------------
// Block-causal conv: 64 blocks per WG. For j=0..7 write emb[k=j]=prefix+bias,
// then (j<7) prefix += A_j @ Wc[j] via WMMA C-operand chaining.
// Wave w owns N-columns [32w,32w+32); 4 row-tiles x 2 col-tiles accumulators.
// Also emits bf16 block-end row (k=7) as next level's source.
// ---------------------------------------------------------------------------
__global__ __launch_bounds__(256) void blockconv_kernel(
    const unsigned short* __restrict__ h,    // (B*L, E) bf16
    const unsigned short* __restrict__ WT,   // (8, E, E) bf16 [j][n][k]
    const float* __restrict__ bias,          // (E)
    unsigned short* __restrict__ emb,        // (B*L, E) bf16
    unsigned short* __restrict__ src)        // (B*M, E) bf16 (may be null)
{
  __shared__ __align__(16) unsigned short As[MT * EE];   // 32KB
  const int wave = threadIdx.x >> 5, lane = threadIdx.x & 31;
  const long blkbase = (long)blockIdx.x * MT;
  const int n0   = wave * 32;
  const int nn   = lane & 15;
  const int half = lane >> 4;
  const int c0   = half * 8;

  v8f acc[4][2];
  #pragma unroll
  for (int mt = 0; mt < 4; ++mt)
    #pragma unroll
    for (int ct = 0; ct < 2; ++ct)
      #pragma unroll
      for (int r = 0; r < 8; ++r) acc[mt][ct][r] = 0.f;

  const float b0 = bias[n0 + nn];
  const float b1 = bias[n0 + 16 + nn];

  for (int j = 0; j < CSZ; ++j) {
    // emit emb rows for k=j (strict-causal prefix, excludes t_j)
    #pragma unroll
    for (int mt = 0; mt < 4; ++mt) {
      #pragma unroll
      for (int r = 0; r < 8; ++r) {
        const int m = mt * 16 + r + half * 8;
        const long token = (blkbase + m) * CSZ + j;
        const unsigned short e0 = f2bf(acc[mt][0][r] + b0);
        const unsigned short e1 = f2bf(acc[mt][1][r] + b1);
        emb[token * EE + n0 + nn]      = e0;
        emb[token * EE + n0 + 16 + nn] = e1;
        if (j == CSZ - 1 && src) {
          src[(blkbase + m) * EE + n0 + nn]      = e0;
          src[(blkbase + m) * EE + n0 + 16 + nn] = e1;
        }
      }
    }
    if (j == CSZ - 1) break;                    // t[7] never needed

    __syncthreads();
    {   // stage A_j: 64 rows (token = (blk, j)); 128B per thread
      const int t  = threadIdx.x;
      const int rr = t >> 2;                    // 4 threads per 512B row
      const int q  = t & 3;
      const uint4* sp = (const uint4*)(h + ((blkbase + rr) * CSZ + j) * EE) + q * 8;
      uint4* dp = (uint4*)(As + rr * EE) + q * 8;
      #pragma unroll
      for (int i = 0; i < 8; ++i) dp[i] = sp[i];
    }
    __syncthreads();

    const unsigned short* Wt = WT + (size_t)j * EE * EE;
    #pragma unroll
    for (int kb = 0; kb < EE / 32; ++kb) {
      Frag bA = load_frag(&Wt[(size_t)(n0 + nn) * EE + kb * 32 + c0]);
      Frag bB = load_frag(&Wt[(size_t)(n0 + 16 + nn) * EE + kb * 32 + c0]);
      #pragma unroll
      for (int mt = 0; mt < 4; ++mt) {
        Frag a = load_frag(&As[(mt * 16 + nn) * EE + kb * 32 + c0]);
        acc[mt][0] = wmma_bf16(a, bA, acc[mt][0]);   // prefix accumulation
        acc[mt][1] = wmma_bf16(a, bB, acc[mt][1]);
      }
    }
  }
}

// ---------------------------------------------------------------------------
// Deconv: y[q*8+k] = x[q] @ Wd[k] + bd + emb. 64 q-rows per WG, A staged once.
// ---------------------------------------------------------------------------
__global__ __launch_bounds__(256) void deconv_kernel(
    const unsigned short* __restrict__ x,    // (B*M, E) bf16
    const unsigned short* __restrict__ WT,   // (8, E, E) bf16 [k][n][kk]
    const float* __restrict__ bias,
    const unsigned short* __restrict__ emb,  // (B*L, E) bf16
    unsigned short* __restrict__ y)          // (B*L, E) bf16
{
  __shared__ __align__(16) unsigned short As[MT * EE];   // 32KB
  const int wave = threadIdx.x >> 5, lane = threadIdx.x & 31;
  const long qbase = (long)blockIdx.x * MT;
  const int n0   = wave * 32;
  const int nn   = lane & 15;
  const int half = lane >> 4;
  const int c0   = half * 8;

  {   // stage 64 contiguous rows of x (32KB), 128B per thread
    const int t = threadIdx.x;
    const uint4* sp = (const uint4*)(x + qbase * EE) + t * 8;
    uint4* dp = (uint4*)As + t * 8;
    #pragma unroll
    for (int i = 0; i < 8; ++i) dp[i] = sp[i];
  }
  __syncthreads();

  const float b0 = bias[n0 + nn];
  const float b1 = bias[n0 + 16 + nn];

  for (int k = 0; k < CSZ; ++k) {
    v8f acc[4][2];
    #pragma unroll
    for (int mt = 0; mt < 4; ++mt)
      #pragma unroll
      for (int ct = 0; ct < 2; ++ct)
        #pragma unroll
        for (int r = 0; r < 8; ++r) acc[mt][ct][r] = 0.f;

    const unsigned short* Wt = WT + (size_t)k * EE * EE;
    #pragma unroll
    for (int kb = 0; kb < EE / 32; ++kb) {
      Frag bA = load_frag(&Wt[(size_t)(n0 + nn) * EE + kb * 32 + c0]);
      Frag bB = load_frag(&Wt[(size_t)(n0 + 16 + nn) * EE + kb * 32 + c0]);
      #pragma unroll
      for (int mt = 0; mt < 4; ++mt) {
        Frag a = load_frag(&As[(mt * 16 + nn) * EE + kb * 32 + c0]);
        acc[mt][0] = wmma_bf16(a, bA, acc[mt][0]);
        acc[mt][1] = wmma_bf16(a, bB, acc[mt][1]);
      }
    }
    #pragma unroll
    for (int mt = 0; mt < 4; ++mt) {
      #pragma unroll
      for (int r = 0; r < 8; ++r) {
        const int m = mt * 16 + r + half * 8;
        const long token = (qbase + m) * CSZ + k;
        const float e0 = bf2f(emb[token * EE + n0 + nn]);
        const float e1 = bf2f(emb[token * EE + n0 + 16 + nn]);
        y[token * EE + n0 + nn]      = f2bf(acc[mt][0][r] + b0 + e0);
        y[token * EE + n0 + 16 + nn] = f2bf(acc[mt][1][r] + b1 + e1);
      }
    }
  }
}

// ---------------------------------------------------------------------------
// Scatter: h[b,t] = mask ? src[b, rank[b,t]] : 0. one wave per row.
// ---------------------------------------------------------------------------
__global__ __launch_bounds__(256) void scatter_kernel(
    const unsigned short* __restrict__ src,  // (B*M, E)
    const int* __restrict__ value, long voff,
    const int* __restrict__ rank,            // (B*L)
    unsigned short* __restrict__ h)          // (B*L, E)
{
  const int wave = threadIdx.x >> 5, lane = threadIdx.x & 31;
  const long row = (long)blockIdx.x * 8 + wave;
  const long b = row / LTOK, t = row % LTOK;
  uint4* dp = (uint4*)(h + row * EE) + lane;
  if (value[b * 3 * LTOK + voff + t] == 2) {
    const long r = rank[row];
    *dp = ((const uint4*)(src + (b * MM + r) * EE))[lane];
  } else {
    *dp = make_uint4(0u, 0u, 0u, 0u);
  }
}

// Compact gather: x[b,k] = y[b, idx[b,k]]. one wave per row.
__global__ __launch_bounds__(256) void gather_kernel(
    const unsigned short* __restrict__ y,    // (B*L, E)
    const int* __restrict__ idx,             // (B*M)
    unsigned short* __restrict__ x)          // (B*M, E)
{
  const int wave = threadIdx.x >> 5, lane = threadIdx.x & 31;
  const long row = (long)blockIdx.x * 8 + wave;
  const long b = row / MM;
  const long t = idx[row];
  ((uint4*)(x + row * EE))[lane] = ((const uint4*)(y + (b * LTOK + t) * EE))[lane];
}

// ---------------------------------------------------------------------------
// Logits: out = y @ Wl + bl  (N=16). one wave per 16-row tile, 8 WMMAs.
// ---------------------------------------------------------------------------
__global__ __launch_bounds__(256) void logits_kernel(
    const unsigned short* __restrict__ y,    // (B*L, E) bf16
    const unsigned short* __restrict__ WlT,  // (16, 256) bf16 [n][k]
    const float* __restrict__ bl,            // (16)
    float* __restrict__ out)                 // (B*L, 16)
{
  const int wave = threadIdx.x >> 5, lane = threadIdx.x & 31;
  const long tile = (long)blockIdx.x * 8 + wave;
  const int nn   = lane & 15;
  const int half = lane >> 4;
  const int c0   = half * 8;

  v8f acc = {0.f,0.f,0.f,0.f,0.f,0.f,0.f,0.f};
  const unsigned short* Arow = y + (tile * 16 + nn) * EE;
  #pragma unroll
  for (int kb = 0; kb < EE / 32; ++kb) {
    Frag a = load_frag(Arow + kb * 32 + c0);
    Frag b = load_frag(WlT + (size_t)nn * EE + kb * 32 + c0);
    acc = wmma_bf16(a, b, acc);
  }
  const float bn = bl[nn];
  #pragma unroll
  for (int r = 0; r < 8; ++r) {
    const int m = r + half * 8;
    out[(tile * 16 + m) * VV + nn] = acc[r] + bn;
  }
}

// ---------------------------------------------------------------------------
extern "C" void kernel_launch(void* const* d_in, const int* in_sizes, int n_in,
                              void* d_out, int out_size, void* d_ws, size_t ws_size,
                              hipStream_t stream) {
  (void)in_sizes; (void)n_in; (void)out_size; (void)ws_size;

  const float* x         = (const float*)d_in[0];
  const float* emb_table = (const float*)d_in[1];
  const float* Wc[3] = {(const float*)d_in[2], (const float*)d_in[4], (const float*)d_in[6]};
  const float* bc[3] = {(const float*)d_in[3], (const float*)d_in[5], (const float*)d_in[7]};
  const float* Wd[3] = {(const float*)d_in[8], (const float*)d_in[10], (const float*)d_in[12]};
  const float* bd[3] = {(const float*)d_in[9], (const float*)d_in[11], (const float*)d_in[13]};
  const float* Wl = (const float*)d_in[14];
  const float* bl = (const float*)d_in[15];
  const int* value = (const int*)d_in[16];
  float* out = (float*)d_out;

  // workspace carve-up (256B aligned)
  char* ws = (char*)d_ws;
  auto alloc = [&](size_t bytes) -> char* {
    char* p = ws; ws += (bytes + 255) & ~(size_t)255; return p;
  };
  const size_t wsz  = (size_t)CSZ * EE * EE * 2;        // one level of weights, bf16
  const size_t rowL = (size_t)BB * LTOK * EE * 2;       // (B*L,E) bf16
  const size_t rowM = (size_t)BB * MM * EE * 2;         // (B*M,E) bf16

  unsigned short* wcT   = (unsigned short*)alloc(3 * wsz);
  unsigned short* wdT   = (unsigned short*)alloc(3 * wsz);
  unsigned short* wlT   = (unsigned short*)alloc((size_t)VV * EE * 2);
  unsigned short* hbuf  = (unsigned short*)alloc(rowL);
  unsigned short* embb[3];
  embb[0] = (unsigned short*)alloc(rowL);
  embb[1] = (unsigned short*)alloc(rowL);
  embb[2] = (unsigned short*)alloc(rowL);
  unsigned short* ybuf  = (unsigned short*)alloc(rowL);
  unsigned short* xbuf  = (unsigned short*)alloc(rowM);
  unsigned short* srcb  = (unsigned short*)alloc(rowM);
  int* rank1 = (int*)alloc((size_t)BB * LTOK * 4);
  int* rank2 = (int*)alloc((size_t)BB * LTOK * 4);
  int* idx1  = (int*)alloc((size_t)BB * MM * 4);
  int* idx2  = (int*)alloc((size_t)BB * MM * 4);

  // --- index/scan ---
  build_index_kernel<<<dim3(BB, 2), 256, 0, stream>>>(value, rank1, idx1, rank2, idx2);

  // --- weight convert+transpose -> bf16 [j][n][k] ---
  for (int l = 0; l < 3; ++l) {
    wtcvt_kernel<<<(CSZ * EE * EE + 255) / 256, 256, 0, stream>>>(
        Wc[l], wcT + (size_t)l * CSZ * EE * EE,
        CSZ, EE, EE, (long)EE * EE, (long)EE, 1L);            // Wc: [j][k][n]
    wtcvt_kernel<<<(CSZ * EE * EE + 255) / 256, 256, 0, stream>>>(
        Wd[l], wdT + (size_t)l * CSZ * EE * EE,
        CSZ, EE, EE, (long)EE, (long)CSZ * EE, 1L);           // Wd: [k_in][j][n]
  }
  wtcvt_kernel<<<(VV * EE + 255) / 256, 256, 0, stream>>>(
      Wl, wlT, 1, EE, VV, 0L, (long)VV, 1L);                  // Wl: [k][n]

  // --- embedding lookup (level 0 input) ---
  embed_kernel<<<(BB * LTOK) / 8, 256, 0, stream>>>(emb_table, value, hbuf);

  // --- emb chain (bottom-up) ---
  blockconv_kernel<<<(BB * MM) / MT, 256, 0, stream>>>(
      hbuf, wcT + 0 * CSZ * EE * EE, bc[0], embb[0], srcb);
  scatter_kernel<<<(BB * LTOK) / 8, 256, 0, stream>>>(srcb, value, (long)LTOK, rank1, hbuf);
  blockconv_kernel<<<(BB * MM) / MT, 256, 0, stream>>>(
      hbuf, wcT + 1 * CSZ * EE * EE, bc[1], embb[1], srcb);
  scatter_kernel<<<(BB * LTOK) / 8, 256, 0, stream>>>(srcb, value, 0L, rank2, hbuf);
  blockconv_kernel<<<(BB * MM) / MT, 256, 0, stream>>>(
      hbuf, wcT + 2 * CSZ * EE * EE, bc[2], embb[2], (unsigned short*)nullptr);

  // --- decode chain (top-down) ---
  cvt_rows_kernel<<<(BB * MM) / 8, 256, 0, stream>>>(x, xbuf);
  deconv_kernel<<<(BB * MM) / MT, 256, 0, stream>>>(
      xbuf, wdT + 2 * CSZ * EE * EE, bd[2], embb[2], ybuf);
  gather_kernel<<<(BB * MM) / 8, 256, 0, stream>>>(ybuf, idx2, xbuf);
  deconv_kernel<<<(BB * MM) / MT, 256, 0, stream>>>(
      xbuf, wdT + 1 * CSZ * EE * EE, bd[1], embb[1], ybuf);
  gather_kernel<<<(BB * MM) / 8, 256, 0, stream>>>(ybuf, idx1, xbuf);
  deconv_kernel<<<(BB * MM) / MT, 256, 0, stream>>>(
      xbuf, wdT + 0 * CSZ * EE * EE, bd[0], embb[0], ybuf);

  // --- logits ---
  logits_kernel<<<(BB * LTOK) / (16 * 8), 256, 0, stream>>>(ybuf, wlT, bl, out);
}
